// Attention_19490561590266
// MI455X (gfx1250) — compile-verified, matrix operand order checked by reading, not run on hardware
//
#include <hip/hip_runtime.h>
#include <hip/hip_bf16.h>
#include <math.h>

// ---------------------------------------------------------------------------
// Problem constants (fixed by the reference)
// ---------------------------------------------------------------------------
constexpr int Bc = 2;      // batch
constexpr int Sc = 2048;   // seq len per batch
constexpr int Hc = 32;     // query heads
constexpr int HKc = 8;     // kv heads
constexpr int Dc = 128;    // head dim
constexpr int Gc = Hc / HKc;
constexpr float SCALEc = 0.08838834764831845f;  // 1/sqrt(128)

constexpr int BM = 128;    // query rows per block (8 waves x 16)
constexpr int BN = 32;     // keys per tile (== WMMA K for PV)
constexpr int NWAVES = 8;

typedef __attribute__((ext_vector_type(16))) _Float16 v16h;
typedef __attribute__((ext_vector_type(8)))  _Float16 v8h;
typedef __attribute__((ext_vector_type(2)))  __fp16   h2;   // cvt_pkrtz result type
typedef __attribute__((ext_vector_type(8)))  float    v8f;
typedef __attribute__((ext_vector_type(4)))  float    v4f;

// Build a v16h A/B fragment from two aligned 8-half LDS runs.
__device__ inline v16h make_frag(const _Float16* lo, const _Float16* hi) {
    v8h a = *(const v8h*)lo;
    v8h b = *(const v8h*)hi;
    v16h r;
#pragma unroll
    for (int i = 0; i < 8; ++i) { r[i] = a[i]; r[8 + i] = b[i]; }
    return r;
}

// ---------------------------------------------------------------------------
// KV-cache: copy old cache then scatter fresh rows by slot_mapping
// ---------------------------------------------------------------------------
__global__ void cache_copy_kernel(const float* __restrict__ kc,
                                  const float* __restrict__ vc,
                                  float* __restrict__ okc,
                                  float* __restrict__ ovc, int n) {
    int i = blockIdx.x * blockDim.x + threadIdx.x;
    if (i < n) { okc[i] = kc[i]; ovc[i] = vc[i]; }
}

__global__ void cache_scatter_kernel(const float* __restrict__ k,
                                     const float* __restrict__ v,
                                     const int* __restrict__ slot,
                                     float* __restrict__ okc,
                                     float* __restrict__ ovc,
                                     int nslots) {
    const int tok = blockIdx.x;
    const int s = slot[tok];
    if (s < 0 || s >= nslots) return;
    const int row = HKc * Dc;  // 1024 floats
    const v4f* ks = (const v4f*)(k + (size_t)tok * row);
    const v4f* vs = (const v4f*)(v + (size_t)tok * row);
    v4f* kd = (v4f*)(okc + (size_t)s * row);
    v4f* vd = (v4f*)(ovc + (size_t)s * row);
    for (int i = threadIdx.x; i < row / 4; i += blockDim.x) {
        kd[i] = ks[i];
        vd[i] = vs[i];
    }
}

// ---------------------------------------------------------------------------
// One 32-key tile: QK^T WMMAs, online softmax, PV WMMAs.
// MASKED only for the <=2 tiles per wave that intersect the causal diagonal;
// the branch is wave-uniform so the hot path carries no mask VALU at all.
// ---------------------------------------------------------------------------
template <bool MASKED>
__device__ __forceinline__ void process_tile(
        int kb, int col, int hl, int rowA,
        const v16h (&aQ)[4], v8f (&acc)[8], float& mrow, float& lrow,
        const _Float16 (*__restrict__ sK)[Dc],
        const _Float16 (*__restrict__ sVt)[BN],
        float (*__restrict__ sSw)[BN]) {
    // ---- S = Q K^T : two 16-col subtiles, 4 K-chunks each ----
    v8f s0, s1;
#pragma unroll
    for (int j = 0; j < 8; ++j) { s0[j] = 0.0f; s1[j] = 0.0f; }
#pragma unroll
    for (int c = 0; c < 4; ++c) {
        const _Float16* kr0 = &sK[col][c * 32];
        const _Float16* kr1 = &sK[16 + col][c * 32];
        v16h bK0 = make_frag(kr0 + 8 * hl, kr0 + 16 + 8 * hl);
        v16h bK1 = make_frag(kr1 + 8 * hl, kr1 + 16 + 8 * hl);
        s0 = __builtin_amdgcn_wmma_f32_16x16x32_f16(
                false, aQ[c], false, bK0, (short)0, s0, false, false);
        s1 = __builtin_amdgcn_wmma_f32_16x16x32_f16(
                false, aQ[c], false, bK1, (short)0, s1, false, false);
    }

    // ---- Re-stripe raw scores C-layout -> A-layout via LDS ----
    // (same-wave LDS ops are in-order; no barrier needed)
#pragma unroll
    for (int j = 0; j < 8; ++j) {
        sSw[j + 8 * hl][col]      = s0[j];
        sSw[j + 8 * hl][16 + col] = s1[j];
    }
    const float* pr = &sSw[col][0];
    v4f r0 = *(const v4f*)(pr + 8 * hl);
    v4f r1 = *(const v4f*)(pr + 8 * hl + 4);
    v4f r2 = *(const v4f*)(pr + 16 + 8 * hl);
    v4f r3 = *(const v4f*)(pr + 16 + 8 * hl + 4);

    // ---- Per-row softmax (row = rowA), almost entirely in VALU ----
    float xv[16];
#pragma unroll
    for (int i = 0; i < 4; ++i) {
        xv[i]      = r0[i] * SCALEc;
        xv[4 + i]  = r1[i] * SCALEc;
        xv[8 + i]  = r2[i] * SCALEc;
        xv[12 + i] = r3[i] * SCALEc;
    }
    if constexpr (MASKED) {
#pragma unroll
        for (int e = 0; e < 16; ++e) {
            const int key = kb + 8 * hl + (e < 8 ? e : e + 8);
            if (key > rowA) xv[e] = -1e30f;
        }
    }
    float mx = xv[0];
#pragma unroll
    for (int e = 1; e < 16; ++e) mx = fmaxf(mx, xv[e]);
    mx = fmaxf(mx, __shfl_xor(mx, 16, 32));      // combine lane halves
    const float mnew  = fmaxf(mrow, mx);
    const float alpha = __expf(mrow - mnew);
    float pe[16];
    float sum = 0.0f;
#pragma unroll
    for (int e = 0; e < 16; ++e) {
        pe[e] = __expf(xv[e] - mnew);
        sum += pe[e];
    }
    v16h aP;
#pragma unroll
    for (int e = 0; e < 8; ++e) {
        h2 pk = __builtin_amdgcn_cvt_pkrtz(pe[2 * e], pe[2 * e + 1]);
        aP[2 * e]     = (_Float16)pk[0];
        aP[2 * e + 1] = (_Float16)pk[1];
    }
    sum += __shfl_xor(sum, 16, 32);              // combine lane halves
    lrow = lrow * alpha + sum;
    mrow = mnew;

    // ---- Rescale accumulators: broadcast alpha back to C-layout ----
#pragma unroll
    for (int j = 0; j < 8; ++j) {
        const float aj = __shfl(alpha, j + 8 * hl, 32);
#pragma unroll
        for (int t8 = 0; t8 < 8; ++t8) acc[t8][j] *= aj;
    }

    // ---- O += P V  (aP already in A-layout registers) ----
#pragma unroll
    for (int t8 = 0; t8 < 8; ++t8) {
        const _Float16* vc = &sVt[t8 * 16 + col][0];
        v16h bV = make_frag(vc + 8 * hl, vc + 16 + 8 * hl);
        acc[t8] = __builtin_amdgcn_wmma_f32_16x16x32_f16(
                false, aP, false, bV, (short)0, acc[t8], false, false);
    }
}

// ---------------------------------------------------------------------------
// Flash attention forward (causal, GQA), f16 WMMA with f32 accumulation
// grid = (S/BM, H, B), block = 256 (8 wave32)
// ---------------------------------------------------------------------------
__global__ __launch_bounds__(256)
void fa_fwd_kernel(const float* __restrict__ q,
                   const float* __restrict__ k,
                   const float* __restrict__ v,
                   float* __restrict__ o) {
    __shared__ _Float16 sK[BN][Dc];            //  8 KB  [key][d]
    __shared__ _Float16 sVt[Dc][BN];           //  8 KB  [d][key] (transposed V)
    __shared__ float    sS[NWAVES][16][BN];    // 16 KB  score re-stripe scratch

    const int q0 = blockIdx.x * BM;
    const int h  = blockIdx.y;
    const int b  = blockIdx.z;
    const int hk = h / Gc;

    const int tid  = threadIdx.x;
    const int lane = tid & 31;
    const int wave = tid >> 5;
    const int col  = lane & 15;   // N (C-layout) / M (A-layout) index
    const int hl   = lane >> 4;   // lane-half select (K split)

    const int qw   = q0 + wave * 16;  // this wave's first query row
    const int rowA = qw + col;        // A-layout query row owned by this lane

    // ---- Load Q fragments: 16 rows x 128 d as 4 A-frags (f32 -> f16) ----
    v16h aQ[4];
    {
        const float* qrow = q + (((size_t)(b * Sc + rowA)) * Hc + h) * Dc;
#pragma unroll
        for (int c = 0; c < 4; ++c) {
            const v4f* p0 = (const v4f*)(qrow + c * 32 + 8 * hl);
            const v4f* p1 = (const v4f*)(qrow + c * 32 + 16 + 8 * hl);
            v4f f0 = p0[0], f1 = p0[1], f2 = p1[0], f3 = p1[1];
            float fr[16];
#pragma unroll
            for (int i = 0; i < 4; ++i) {
                fr[i] = f0[i]; fr[4 + i] = f1[i];
                fr[8 + i] = f2[i]; fr[12 + i] = f3[i];
            }
            v16h a;
#pragma unroll
            for (int i = 0; i < 8; ++i) {
                h2 pk = __builtin_amdgcn_cvt_pkrtz(fr[2 * i], fr[2 * i + 1]);
                a[2 * i]     = (_Float16)pk[0];
                a[2 * i + 1] = (_Float16)pk[1];
            }
            aQ[c] = a;
        }
    }

    // ---- Accumulators (C-layout) & per-row softmax state (A-layout) ----
    v8f acc[8];
#pragma unroll
    for (int t = 0; t < 8; ++t)
#pragma unroll
        for (int j = 0; j < 8; ++j) acc[t][j] = 0.0f;
    float mrow = -1e30f;
    float lrow = 0.0f;

    const int ktiles = (q0 + BM) / BN;  // causal bound for whole block
    for (int kt = 0; kt < ktiles; ++kt) {
        const int kb = kt * BN;

        __syncthreads();  // finish reads of previous tile
        // ---- Cooperative stage of K (row-major) and V (transposed) ----
        {
            const int key = tid >> 3;          // 0..31
            const int d0  = (tid & 7) * 16;    // 0..112
            const size_t base = (((size_t)(b * Sc + kb + key)) * HKc + hk) * Dc + d0;
            const v4f* kg = (const v4f*)(k + base);
            const v4f* vg = (const v4f*)(v + base);
            v4f kf[4] = {kg[0], kg[1], kg[2], kg[3]};
            v4f vf[4] = {vg[0], vg[1], vg[2], vg[3]};
            float kr[16];
#pragma unroll
            for (int i = 0; i < 4; ++i) {
                kr[i] = kf[0][i]; kr[4 + i] = kf[1][i];
                kr[8 + i] = kf[2][i]; kr[12 + i] = kf[3][i];
            }
            v8h h0, h1;
#pragma unroll
            for (int i = 0; i < 4; ++i) {
                h2 a = __builtin_amdgcn_cvt_pkrtz(kr[2 * i], kr[2 * i + 1]);
                h2 c = __builtin_amdgcn_cvt_pkrtz(kr[8 + 2 * i], kr[9 + 2 * i]);
                h0[2 * i] = (_Float16)a[0]; h0[2 * i + 1] = (_Float16)a[1];
                h1[2 * i] = (_Float16)c[0]; h1[2 * i + 1] = (_Float16)c[1];
            }
            *(v8h*)&sK[key][d0]     = h0;
            *(v8h*)&sK[key][d0 + 8] = h1;
#pragma unroll
            for (int g = 0; g < 4; ++g)
#pragma unroll
                for (int i = 0; i < 4; ++i)
                    sVt[d0 + g * 4 + i][key] = (_Float16)vf[g][i];
        }
        __syncthreads();

        if (kb <= qw + 15) {  // tile not fully masked for this wave
            if (kb + BN <= qw + 1) {
                process_tile<false>(kb, col, hl, rowA, aQ, acc, mrow, lrow,
                                    sK, sVt, sS[wave]);
            } else {
                process_tile<true>(kb, col, hl, rowA, aQ, acc, mrow, lrow,
                                   sK, sVt, sS[wave]);
            }
        }
    }

    // ---- Finalize: divide by row sums (broadcast to C-layout), store O ----
    const float inv = lrow > 0.0f ? 1.0f / lrow : 0.0f;
#pragma unroll
    for (int j = 0; j < 8; ++j) {
        const float ij = __shfl(inv, j + 8 * hl, 32);
        const int row_g = qw + j + 8 * hl;
        float* orow = o + (((size_t)(b * Sc + row_g)) * Hc + h) * Dc;
#pragma unroll
        for (int t8 = 0; t8 < 8; ++t8)
            orow[t8 * 16 + col] = acc[t8][j] * ij;
    }
}

// ---------------------------------------------------------------------------
// Harness entry
// ---------------------------------------------------------------------------
extern "C" void kernel_launch(void* const* d_in, const int* in_sizes, int n_in,
                              void* d_out, int out_size, void* d_ws, size_t ws_size,
                              hipStream_t stream) {
    const float* q  = (const float*)d_in[0];
    const float* k  = (const float*)d_in[1];
    const float* v  = (const float*)d_in[2];
    const float* kc = (const float*)d_in[3];
    const float* vc = (const float*)d_in[4];
    const int* slot = (const int*)d_in[5];

    const int N = Bc * Sc;
    float* o   = (float*)d_out;
    float* okc = o + (size_t)N * Hc * Dc;
    float* ovc = okc + (size_t)N * HKc * Dc;

    // KV cache: copy old contents, then scatter fresh K/V rows.
    const int cacheElems = N * HKc * Dc;
    cache_copy_kernel<<<(cacheElems + 255) / 256, 256, 0, stream>>>(
        kc, vc, okc, ovc, cacheElems);
    cache_scatter_kernel<<<N, 256, 0, stream>>>(k, v, slot, okc, ovc, N);

    // Causal GQA prefill attention.
    dim3 grid(Sc / BM, Hc, Bc);
    fa_fwd_kernel<<<grid, 256, 0, stream>>>(q, k, v, o);
}
